// FusedBlock_4964982194665
// MI455X (gfx1250) — compile-verified, matrix operand order checked by reading, not run on hardware
//
#include <hip/hip_runtime.h>
#include <cmath>

// ---------------------------------------------------------------------------
// Problem constants (from the reference): S=2048, B=2, H=2048, NH=16, HD=128
// ---------------------------------------------------------------------------
#define S_   2048
#define B_   2
#define H_   2048
#define NH_  16
#define HD_  128
#define TOK_ (S_ * B_)   // 4096 tokens

typedef __attribute__((ext_vector_type(16))) __bf16 v16bf;
typedef __attribute__((ext_vector_type(8)))  __bf16 v8bf;
typedef __attribute__((ext_vector_type(8)))  float  v8f;

static __device__ __forceinline__ v16bf cat8(v8bf lo, v8bf hi) {
  return __builtin_shufflevector(lo, hi, 0,1,2,3,4,5,6,7,8,9,10,11,12,13,14,15);
}

// Load a 16-element bf16 fragment per lane: two contiguous 16B chunks at
// (ptr) and (ptr+16).  Caller folds in row, k-base and lane koff (ISA 7.12.2
// layout for 16-bit A/B matrices, wave32).
static __device__ __forceinline__ v16bf ldfrag(const __bf16* p) {
  v8bf lo = *(const v8bf*)(p);
  v8bf hi = *(const v8bf*)(p + 16);
  return cat8(lo, hi);
}

// CDNA5 async global->LDS copy (16B per lane), tracked by ASYNCcnt.
// VDST VGPR holds the LDS byte offset relative to LDS_BASE (ISA 10.2: the
// generic-pointer low 32 bits are exactly that offset for shared objects).
static __device__ __forceinline__ void async_ld16(unsigned lds_off, const __bf16* g) {
  asm volatile("global_load_async_to_lds_b128 %0, %1, off"
               :: "v"(lds_off), "v"(g) : "memory");
}
static __device__ __forceinline__ void wait_async0() {
  asm volatile("s_wait_asynccnt 0x0" ::: "memory");
}

// ---------------------------------------------------------------------------
// f32 -> bf16 conversion (weights, once per launch; ~13us of HBM time total)
// ---------------------------------------------------------------------------
__global__ void f32_to_bf16_kernel(const float* __restrict__ in,
                                   __bf16* __restrict__ out, size_t n) {
  size_t i = (size_t)blockIdx.x * blockDim.x + threadIdx.x;
  const size_t stride = (size_t)gridDim.x * blockDim.x;
  for (; i < n; i += stride) out[i] = (__bf16)in[i];
}

// ---------------------------------------------------------------------------
// LayerNorm over last axis (H=2048), one row per block, output bf16
// ---------------------------------------------------------------------------
__global__ __launch_bounds__(256)
void ln_bf16_kernel(const float* __restrict__ x, const float* __restrict__ w,
                    const float* __restrict__ b, __bf16* __restrict__ out) {
  __shared__ float red_s[8], red_ss[8];
  const int row = blockIdx.x;
  const float* xr = x + (size_t)row * H_;
  float vals[8];
  float s = 0.f, ss = 0.f;
#pragma unroll
  for (int i = 0; i < 8; ++i) {
    float v = xr[threadIdx.x + i * 256];
    vals[i] = v; s += v; ss += v * v;
  }
#pragma unroll
  for (int off = 16; off > 0; off >>= 1) {
    s  += __shfl_xor(s,  off, 32);
    ss += __shfl_xor(ss, off, 32);
  }
  const int lane = threadIdx.x & 31, wv = threadIdx.x >> 5;
  if (lane == 0) { red_s[wv] = s; red_ss[wv] = ss; }
  __syncthreads();
  s = 0.f; ss = 0.f;
#pragma unroll
  for (int i = 0; i < 8; ++i) { s += red_s[i]; ss += red_ss[i]; }
  const float mu  = s * (1.0f / H_);
  const float var = ss * (1.0f / H_) - mu * mu;
  const float rs  = rsqrtf(var + 1e-5f);
#pragma unroll
  for (int i = 0; i < 8; ++i) {
    const int c = threadIdx.x + i * 256;
    out[(size_t)row * H_ + c] = (__bf16)((vals[i] - mu) * rs * w[c] + b[c]);
  }
}

// ---------------------------------------------------------------------------
// Tiled bf16 GEMM:  C[M,N] = A[M,K] * W[N,K]^T + bias (+epilogue)
// 128x128 block tile, K-step 32, 8 waves, each wave: 32x64 patch (8 WMMAs/step)
// Tiles staged with CDNA5 async global->LDS copies (ASYNCcnt), double-buffered
// LDS with padded stride 40 elems (80B) to spread banks.
// EPI: 0 = scatter into q/k/v^T bf16 buffers (fused QKV split+transpose)
//      1 = +residual -> f32 out
//      2 = exact GELU -> bf16 out
// ---------------------------------------------------------------------------
#define BM  128
#define BN  128
#define BK  32
#define LDP 40

template <int EPI>
__global__ __launch_bounds__(256)
void gemm_bf16(const __bf16* __restrict__ A, const __bf16* __restrict__ W,
               const float* __restrict__ bias, const float* __restrict__ resid,
               float* __restrict__ outF, __bf16* __restrict__ outB,
               __bf16* __restrict__ qout, __bf16* __restrict__ kout,
               __bf16* __restrict__ vtout, int M, int N, int K) {
  __shared__ __bf16 As[2][BM * LDP];
  __shared__ __bf16 Bs[2][BM * LDP];
  const int tid  = threadIdx.x;
  const int lane = tid & 31;
  const int wave = tid >> 5;
  const int hi   = lane >> 4;       // lane half (0/1)
  const int l16  = lane & 15;
  const int koff = hi * 8;          // ISA 7.12.2 16-bit A/B lane K-offset
  const int wm   = wave >> 1;       // 0..3 -> 32-row strips
  const int wn   = wave & 1;        // 0..1 -> 64-col strips
  const int m0   = blockIdx.y * BM;
  const int n0   = blockIdx.x * BN;

  // cooperative tile loaders: 256 threads x 16 bf16 = 128x32 tile per matrix
  const int lrow = tid >> 1;
  const int lcol = (tid & 1) * 16;
  const __bf16* Aload = A + (size_t)(m0 + lrow) * K + lcol;
  const __bf16* Wload = W + (size_t)(n0 + lrow) * K + lcol;
  const unsigned aoff[2] = {
      (unsigned)(uintptr_t)&As[0][lrow * LDP + lcol],
      (unsigned)(uintptr_t)&As[1][lrow * LDP + lcol]};
  const unsigned boff[2] = {
      (unsigned)(uintptr_t)&Bs[0][lrow * LDP + lcol],
      (unsigned)(uintptr_t)&Bs[1][lrow * LDP + lcol]};

  v8f acc[2][4];
#pragma unroll
  for (int r = 0; r < 2; ++r)
#pragma unroll
    for (int c = 0; c < 4; ++c) acc[r][c] = (v8f){0,0,0,0,0,0,0,0};

  const int nK = K / BK;

  // async prologue fill of buffer 0
  {
    async_ld16(aoff[0],      Aload);
    async_ld16(aoff[0] + 16, Aload + 8);
    async_ld16(boff[0],      Wload);
    async_ld16(boff[0] + 16, Wload + 8);
    wait_async0();
    __syncthreads();
  }

  for (int kt = 0; kt < nK; ++kt) {
    const int cur = kt & 1;
    if (kt + 1 < nK) {  // async fill of the other buffer while we compute
      const int nxt = cur ^ 1;
      const __bf16* Ap = Aload + (size_t)(kt + 1) * BK;
      const __bf16* Wp = Wload + (size_t)(kt + 1) * BK;
      async_ld16(aoff[nxt],      Ap);
      async_ld16(aoff[nxt] + 16, Ap + 8);
      async_ld16(boff[nxt],      Wp);
      async_ld16(boff[nxt] + 16, Wp + 8);
    }
    if (kt + 2 < nK) {  // L2 prefetch of next-next tile
      __builtin_prefetch(Aload + (size_t)(kt + 2) * BK, 0, 1);
      __builtin_prefetch(Wload + (size_t)(kt + 2) * BK, 0, 1);
    }

    v16bf af[2], bf[4];
#pragma unroll
    for (int r = 0; r < 2; ++r)
      af[r] = ldfrag(&As[cur][(wm * 32 + r * 16 + l16) * LDP + koff]);
#pragma unroll
    for (int c = 0; c < 4; ++c)
      bf[c] = ldfrag(&Bs[cur][(wn * 64 + c * 16 + l16) * LDP + koff]);

#pragma unroll
    for (int r = 0; r < 2; ++r)
#pragma unroll
      for (int c = 0; c < 4; ++c)
        acc[r][c] = __builtin_amdgcn_wmma_f32_16x16x32_bf16(
            false, af[r], false, bf[c], (short)0, acc[r][c], false, false);

    wait_async0();      // my wave's async fill of nxt has landed in LDS
    __syncthreads();    // everyone's has; nxt becomes cur
  }

  // epilogue: C lane mapping M = vgpr + 8*hi, N = l16 (ISA 7.12.2)
#pragma unroll
  for (int r = 0; r < 2; ++r) {
#pragma unroll
    for (int c = 0; c < 4; ++c) {
      const int col = n0 + wn * 64 + c * 16 + l16;
      const float bv = bias[col];
#pragma unroll
      for (int i = 0; i < 8; ++i) {
        const int row = m0 + wm * 32 + r * 16 + hi * 8 + i;
        float v = acc[r][c][i] + bv;
        if (EPI == 0) {
          // fused QKV split + V transpose:  row = s*B + b, col = head*384 + w
          const int b_ = row % B_, s_ = row / B_;
          const int head = col / (3 * HD_);
          const int w_   = col - head * (3 * HD_);
          if (w_ < HD_)
            qout[((size_t)(b_ * NH_ + head) * S_ + s_) * HD_ + w_] = (__bf16)v;
          else if (w_ < 2 * HD_)
            kout[((size_t)(b_ * NH_ + head) * S_ + s_) * HD_ + (w_ - HD_)] = (__bf16)v;
          else
            vtout[((size_t)(b_ * NH_ + head) * HD_ + (w_ - 2 * HD_)) * S_ + s_] = (__bf16)v;
        } else if (EPI == 1) {
          v += resid[(size_t)row * N + col];
          outF[(size_t)row * N + col] = v;
        } else {
          v = 0.5f * v * (1.0f + erff(v * 0.70710678118654752f));
          outB[(size_t)row * N + col] = (__bf16)v;
        }
      }
    }
  }
}

// ---------------------------------------------------------------------------
// Flash attention: one wave owns 16 query rows, streams 32-key blocks.
// Q.K^T via 8 WMMAs, online softmax (16-lane shfl reductions match the C
// layout), P staged through per-wave LDS (s_wait_dscnt), P.V via 8 WMMAs
// against pre-transposed V^T.  Output written bf16 token-major for proj GEMM.
// ---------------------------------------------------------------------------
__global__ __launch_bounds__(256)
void flash_attn_kernel(const __bf16* __restrict__ q, const __bf16* __restrict__ k,
                       const __bf16* __restrict__ vt, __bf16* __restrict__ out) {
  __shared__ __bf16 Psm[8][16 * LDP];   // per-wave P tile (16x32, padded)
  const int lane = threadIdx.x & 31;
  const int wave = threadIdx.x >> 5;
  const int hi = lane >> 4, l16 = lane & 15, koff = hi * 8;
  const int b = blockIdx.z, h = blockIdx.y;
  const int qbase = blockIdx.x * 128 + wave * 16;
  const __bf16* qp = q  + (size_t)(b * NH_ + h) * S_ * HD_;
  const __bf16* kp = k  + (size_t)(b * NH_ + h) * S_ * HD_;
  const __bf16* vp = vt + (size_t)(b * NH_ + h) * HD_ * S_;
  __bf16* Pw = &Psm[wave][0];

  v16bf qf[4];
#pragma unroll
  for (int c = 0; c < 4; ++c)
    qf[c] = ldfrag(qp + (size_t)(qbase + l16) * HD_ + c * 32 + koff);

  v8f acc[8];
#pragma unroll
  for (int j = 0; j < 8; ++j) acc[j] = (v8f){0,0,0,0,0,0,0,0};
  float mrow[8], lrow[8];
#pragma unroll
  for (int i = 0; i < 8; ++i) { mrow[i] = -1e30f; lrow[i] = 0.f; }
  const float sm = 0.08838834764831845f;  // 1/sqrt(HD)

  for (int kb = 0; kb < S_ / 32; ++kb) {
    v8f s0 = (v8f){0,0,0,0,0,0,0,0};
    v8f s1 = (v8f){0,0,0,0,0,0,0,0};
#pragma unroll
    for (int c = 0; c < 4; ++c) {
      const __bf16* p0 = kp + (size_t)(kb * 32 + l16) * HD_ + c * 32 + koff;
      v16bf kf0 = ldfrag(p0);
      v16bf kf1 = ldfrag(p0 + 16 * HD_);
      s0 = __builtin_amdgcn_wmma_f32_16x16x32_bf16(false, qf[c], false, kf0,
                                                   (short)0, s0, false, false);
      s1 = __builtin_amdgcn_wmma_f32_16x16x32_bf16(false, qf[c], false, kf1,
                                                   (short)0, s1, false, false);
    }
#pragma unroll
    for (int i = 0; i < 8; ++i) {
      const float v0 = s0[i] * sm, v1 = s1[i] * sm;
      float r = fmaxf(v0, v1);
      r = fmaxf(r, __shfl_xor(r, 1, 32));
      r = fmaxf(r, __shfl_xor(r, 2, 32));
      r = fmaxf(r, __shfl_xor(r, 4, 32));
      r = fmaxf(r, __shfl_xor(r, 8, 32));
      const float mn = fmaxf(mrow[i], r);
      const float a  = __expf(mrow[i] - mn);
      const float p0v = __expf(v0 - mn);
      const float p1v = __expf(v1 - mn);
      float rs = p0v + p1v;
      rs += __shfl_xor(rs, 1, 32);
      rs += __shfl_xor(rs, 2, 32);
      rs += __shfl_xor(rs, 4, 32);
      rs += __shfl_xor(rs, 8, 32);
      lrow[i] = lrow[i] * a + rs;
      mrow[i] = mn;
#pragma unroll
      for (int j = 0; j < 8; ++j) acc[j][i] *= a;
      const int prow = i + 8 * hi;
      Pw[prow * LDP + l16]      = (__bf16)p0v;
      Pw[prow * LDP + 16 + l16] = (__bf16)p1v;
    }
    // make cross-lane LDS writes visible before re-reading as an A fragment
    asm volatile("s_wait_dscnt 0" ::: "memory");
    v16bf pf = ldfrag(&Pw[l16 * LDP + koff]);
#pragma unroll
    for (int j = 0; j < 8; ++j) {
      v16bf vf = ldfrag(vp + (size_t)(j * 16 + l16) * S_ + kb * 32 + koff);
      acc[j] = __builtin_amdgcn_wmma_f32_16x16x32_bf16(false, pf, false, vf,
                                                       (short)0, acc[j], false, false);
    }
  }
  // out[s, b, head*HD + d] as bf16 (input of the proj GEMM)
#pragma unroll
  for (int j = 0; j < 8; ++j) {
    const int col = h * HD_ + j * 16 + l16;
#pragma unroll
    for (int i = 0; i < 8; ++i) {
      const int qrow = qbase + i + 8 * hi;
      out[((size_t)qrow * B_ + b) * H_ + col] = (__bf16)(acc[j][i] / lrow[i]);
    }
  }
}

// ---------------------------------------------------------------------------
// Host-side orchestration
// ---------------------------------------------------------------------------
extern "C" void kernel_launch(void* const* d_in, const int* in_sizes, int n_in,
                              void* d_out, int out_size, void* d_ws, size_t ws_size,
                              hipStream_t stream) {
  const float* x     = (const float*)d_in[0];
  const float* ln1w  = (const float*)d_in[1];
  const float* ln1b  = (const float*)d_in[2];
  const float* Wqkv  = (const float*)d_in[3];
  const float* bqkv  = (const float*)d_in[4];
  const float* Wproj = (const float*)d_in[5];
  const float* bproj = (const float*)d_in[6];
  const float* ln2w  = (const float*)d_in[7];
  const float* ln2b  = (const float*)d_in[8];
  const float* Wfc1  = (const float*)d_in[9];
  const float* bfc1  = (const float*)d_in[10];
  const float* Wfc2  = (const float*)d_in[11];
  const float* bfc2  = (const float*)d_in[12];
  float* out = (float*)d_out;

  // workspace carve-up (256B aligned)
  char* ws = (char*)d_ws;
  size_t off = 0;
  auto carve = [&](size_t bytes) {
    void* p = ws + off;
    off = (off + bytes + 255) & ~(size_t)255;
    return p;
  };
  const size_t nWqkv = (size_t)3 * H_ * H_;
  const size_t nWpr  = (size_t)H_ * H_;
  const size_t nWf1  = (size_t)4 * H_ * H_;
  const size_t nWf2  = (size_t)H_ * 4 * H_;
  const size_t nTokH = (size_t)TOK_ * H_;
  const size_t nHead = (size_t)B_ * NH_ * S_ * HD_;   // == TOK_*H_

  __bf16* wqkv_bf = (__bf16*)carve(nWqkv * 2);
  __bf16* wproj_bf = (__bf16*)carve(nWpr * 2);
  __bf16* wfc1_bf = (__bf16*)carve(nWf1 * 2);
  __bf16* wfc2_bf = (__bf16*)carve(nWf2 * 2);
  __bf16* xln   = (__bf16*)carve(nTokH * 2);
  __bf16* qb    = (__bf16*)carve(nHead * 2);
  __bf16* kb    = (__bf16*)carve(nHead * 2);
  __bf16* vtb   = (__bf16*)carve(nHead * 2);
  __bf16* attnb = (__bf16*)carve(nTokH * 2);
  float*  x1    = (float*)carve(nTokH * 4);
  __bf16* x2ln  = (__bf16*)carve(nTokH * 2);
  __bf16* hbuf  = (__bf16*)carve((size_t)TOK_ * 4 * H_ * 2);

  // 1) weight conversions f32 -> bf16
  f32_to_bf16_kernel<<<4096, 256, 0, stream>>>(Wqkv, wqkv_bf, nWqkv);
  f32_to_bf16_kernel<<<4096, 256, 0, stream>>>(Wproj, wproj_bf, nWpr);
  f32_to_bf16_kernel<<<4096, 256, 0, stream>>>(Wfc1, wfc1_bf, nWf1);
  f32_to_bf16_kernel<<<4096, 256, 0, stream>>>(Wfc2, wfc2_bf, nWf2);

  // 2) LN1 -> bf16
  ln_bf16_kernel<<<TOK_, 256, 0, stream>>>(x, ln1w, ln1b, xln);

  // 3) QKV GEMM with fused split + V-transpose epilogue (M=4096, N=6144, K=2048)
  gemm_bf16<0><<<dim3(6144 / BN, TOK_ / BM), 256, 0, stream>>>(
      xln, wqkv_bf, bqkv, nullptr, nullptr, nullptr, qb, kb, vtb,
      TOK_, 3 * H_, H_);

  // 4) flash attention -> bf16 [token, H]
  flash_attn_kernel<<<dim3(S_ / 128, NH_, B_), 256, 0, stream>>>(qb, kb, vtb, attnb);

  // 5) proj GEMM + residual(x) -> x1 f32 (M=4096, N=2048, K=2048)
  gemm_bf16<1><<<dim3(H_ / BN, TOK_ / BM), 256, 0, stream>>>(
      attnb, wproj_bf, bproj, x, x1, nullptr, nullptr, nullptr, nullptr,
      TOK_, H_, H_);

  // 6) LN2 -> bf16
  ln_bf16_kernel<<<TOK_, 256, 0, stream>>>(x1, ln2w, ln2b, x2ln);

  // 7) FC1 GEMM + exact GELU -> bf16 (M=4096, N=8192, K=2048)
  gemm_bf16<2><<<dim3(4 * H_ / BN, TOK_ / BM), 256, 0, stream>>>(
      x2ln, wfc1_bf, bfc1, nullptr, nullptr, hbuf, nullptr, nullptr, nullptr,
      TOK_, 4 * H_, H_);

  // 8) FC2 GEMM + residual(x1) -> d_out f32 (M=4096, N=2048, K=8192)
  gemm_bf16<1><<<dim3(H_ / BN, TOK_ / BM), 256, 0, stream>>>(
      hbuf, wfc2_bf, bfc2, x1, out, nullptr, nullptr, nullptr, nullptr,
      TOK_, H_, 4 * H_);

  (void)in_sizes; (void)n_in; (void)out_size; (void)ws_size;
}
